// RGBuvHistBlock_10952166604937
// MI455X (gfx1250) — matrix operand order repaired and checked by reading, not compiled, not run on hardware
//
#include <hip/hip_runtime.h>

typedef __attribute__((ext_vector_type(16))) _Float16 v16h;
typedef __attribute__((ext_vector_type(8)))  _Float16 v8h;
typedef __attribute__((ext_vector_type(8)))  float    v8f;

#define NB     8
#define NC     3
#define NPIX   65536
#define HBINS  64
#define EPS    1e-6f
#define INV_S2 2500.0f              // 1/sigma^2, sigma = 0.02
#define LN2    0.69314718055994531f
#define STEP   (6.0f / 63.0f)       // linspace(-3,3,64) step
#define PHASE  256                  // pixels staged in LDS per phase
#define ROUND  128                  // pixels per produce/consume round (4 K-chunks)
#define CHUNKS 16                   // N-splits per (b,c)
#define PIXPB  (NPIX / CHUNKS)      // 4096 pixels per block

#define SHUF16(lo, hi) __builtin_shufflevector(lo, hi, 0,1,2,3,4,5,6,7,8,9,10,11,12,13,14,15)

// ---------------------------------------------------------------------------
// Zero the output accumulator (harness poisons d_out with 0xAA).
// ---------------------------------------------------------------------------
__global__ void zero_kernel(float* __restrict__ out, int n) {
    int i = blockIdx.x * blockDim.x + threadIdx.x;
    if (i < n) out[i] = 0.0f;
}

// ---------------------------------------------------------------------------
// Histogram accumulation via v_wmma_f32_16x16x32_f16 with LDS fragment
// sharing: 8 unique fragments per 32-pixel K-chunk (4 A rows, 4 B cols) are
// produced once each by the 8 waves, then consumed by all waves. This cuts
// v_rcp_f32 (TRANS, contends with WMMA) work 3x vs. recompute-per-wave.
// ---------------------------------------------------------------------------
__global__ __launch_bounds__(256)
void hist_kernel(const float* __restrict__ x, float* __restrict__ out) {
    __shared__ float su[PHASE];          // log-chroma u per staged pixel
    __shared__ float sv[PHASE];          // log-chroma v per staged pixel
    __shared__ float sy[PHASE];          // intensity scale Iy per staged pixel
    // frag[fragid][chunk][lane][2 x v8h + 1 pad] : fragid 0-3 = A rows, 4-7 = B cols
    __shared__ v8h frag[8][4][32][3];    // 48 KB

    const int bid   = blockIdx.x;
    const int chunk = bid % CHUNKS;
    const int bc    = bid / CHUNKS;
    const int b     = bc / NC;
    const int ch    = bc % NC;

    const int tid  = threadIdx.x;
    const int lane = tid & 31;
    const int wave = tid >> 5;
    const int hh   = lane >> 4;          // half-wave: 0 or 1
    const int l15  = lane & 15;

    // Tile assignment: wave w owns tiles 2w and 2w+1 of the 4x4 tile grid.
    const int tile0 = wave * 2;
    const int r  = tile0 >> 2;           // u-bin tile row (shared by both tiles)
    const int c0 = tile0 & 3;            // v-bin tile cols
    const int c1 = c0 + 1;

    // Producer role: waves 0-3 build A fragment for row r=wave,
    //                waves 4-7 build B fragment for col c=wave-4.
    const bool isA   = wave < 4;
    const float binP = -3.0f + STEP * (float)((isA ? wave : wave - 4) * 16 + l15);

    const float* xb   = x + (size_t)b * NC * NPIX;
    const int    base = chunk * PIXPB;

    v8f acc0 = {};
    v8f acc1 = {};

    for (int p0 = 0; p0 < PIXPB; p0 += PHASE) {
        // ---- Stage PHASE pixels: per-pixel transcendental work done once ----
        {
            const int pix = base + p0 + tid;
            float rr = xb[0 * NPIX + pix];
            float gg = xb[1 * NPIX + pix];
            float bb = xb[2 * NPIX + pix];
            rr = fminf(fmaxf(rr * 0.5f + 0.5f, 0.0f), 1.0f);
            gg = fminf(fmaxf(gg * 0.5f + 0.5f, 0.0f), 1.0f);
            bb = fminf(fmaxf(bb * 0.5f + 0.5f, 0.0f), 1.0f);
            float iy  = __builtin_amdgcn_sqrtf(rr * rr + gg * gg + bb * bb + EPS);
            float l2r = __builtin_amdgcn_logf(rr + EPS);   // log2
            float l2g = __builtin_amdgcn_logf(gg + EPS);
            float l2b = __builtin_amdgcn_logf(bb + EPS);
            // Branch-free channel select (ch is block-uniform -> cndmask)
            float la = (ch == 0) ? l2r : ((ch == 1) ? l2g : l2b);
            float lu = (ch == 0) ? l2g : l2r;
            float lv = (ch == 2) ? l2g : l2b;
            su[tid] = (la - lu) * LN2;
            sv[tid] = (la - lv) * LN2;
            sy[tid] = iy;
        }
        __syncthreads();

        for (int rd = 0; rd < PHASE; rd += ROUND) {
            // ---- Produce: each wave builds its unique fragment, 4 chunks ----
            if (isA) {
                #pragma unroll
                for (int cc = 0; cc < 4; ++cc) {
                    v8h lo, hi;
                    #pragma unroll
                    for (int j = 0; j < 16; ++j) {
                        // A-matrix (16x32 f16) lane/K layout:
                        // lanes 0-15: K={0..7,16..23}; lanes 16-31: K={8..15,24..31}
                        const int ka = rd + cc * 32 + ((j < 8) ? j : j + 8) + hh * 8;
                        float du = su[ka] - binP;
                        float v  = sy[ka] *
                                   __builtin_amdgcn_rcpf(fmaf(du * du, INV_S2, 1.0f));
                        if (j < 8) lo[j] = (_Float16)v; else hi[j - 8] = (_Float16)v;
                    }
                    frag[wave][cc][lane][0] = lo;
                    frag[wave][cc][lane][1] = hi;
                }
            } else {
                #pragma unroll
                for (int cc = 0; cc < 4; ++cc) {
                    v8h lo, hi;
                    #pragma unroll
                    for (int j = 0; j < 16; ++j) {
                        // B-matrix (32x16 f16): lanes 0-15 K={0..7,16..23}; lanes 16-31 K={8..15,24..31}
                        const int kb = rd + cc * 32 + ((j < 8) ? j : j + 8) + hh * 8;
                        float dv = sv[kb] - binP;
                        float v  = __builtin_amdgcn_rcpf(fmaf(dv * dv, INV_S2, 1.0f));
                        if (j < 8) lo[j] = (_Float16)v; else hi[j - 8] = (_Float16)v;
                    }
                    frag[wave][cc][lane][0] = lo;
                    frag[wave][cc][lane][1] = hi;
                }
            }
            __syncthreads();

            // ---- Consume: every wave reads its 3 fragments, 2 WMMAs/chunk ----
            #pragma unroll
            for (int cc = 0; cc < 4; ++cc) {
                v8h alo  = frag[r][cc][lane][0];
                v8h ahi  = frag[r][cc][lane][1];
                v8h b0lo = frag[4 + c0][cc][lane][0];
                v8h b0hi = frag[4 + c0][cc][lane][1];
                v8h b1lo = frag[4 + c1][cc][lane][0];
                v8h b1hi = frag[4 + c1][cc][lane][1];
                v16h fa  = SHUF16(alo,  ahi);
                v16h fb0 = SHUF16(b0lo, b0hi);
                v16h fb1 = SHUF16(b1lo, b1hi);
                acc0 = __builtin_amdgcn_wmma_f32_16x16x32_f16(
                    false, fa, false, fb0, (short)0, acc0, false, false);
                acc1 = __builtin_amdgcn_wmma_f32_16x16x32_f16(
                    false, fa, false, fb1, (short)0, acc1, false, false);
            }
            __syncthreads();
        }
    }

    // ---- Merge the 16 chunk-partials per (b,c) into global accumulator ----
    // C/D f32 16x16 layout: lanes 0-15 N=lane, VGPR j -> M=j; lanes 16-31 M=j+8.
    float* o = out + (size_t)(b * NC + ch) * HBINS * HBINS;
    #pragma unroll
    for (int j = 0; j < 8; ++j) {
        const int u = r * 16 + j + 8 * hh;
        atomicAdd(&o[u * HBINS + c0 * 16 + l15], acc0[j]);
        atomicAdd(&o[u * HBINS + c1 * 16 + l15], acc1[j]);
    }
}

// ---------------------------------------------------------------------------
// Per-batch normalization: hist / (sum(hist over c,h,g) + EPS), in place.
// One block per batch.
// ---------------------------------------------------------------------------
__global__ __launch_bounds__(256)
void norm_kernel(float* __restrict__ out) {
    __shared__ float red[256];
    const int b   = blockIdx.x;
    const int tid = threadIdx.x;
    float* o = out + (size_t)b * NC * HBINS * HBINS;
    const int n = NC * HBINS * HBINS;   // 12288

    float s = 0.0f;
    for (int i = tid; i < n; i += 256) s += o[i];
    red[tid] = s;
    __syncthreads();
    for (int k = 128; k > 0; k >>= 1) {
        if (tid < k) red[tid] += red[tid + k];
        __syncthreads();
    }
    const float inv = 1.0f / (red[0] + EPS);
    for (int i = tid; i < n; i += 256) o[i] *= inv;
}

// ---------------------------------------------------------------------------
extern "C" void kernel_launch(void* const* d_in, const int* in_sizes, int n_in,
                              void* d_out, int out_size, void* d_ws, size_t ws_size,
                              hipStream_t stream) {
    const float* x   = (const float*)d_in[0];
    float*       out = (float*)d_out;

    const int n = NB * NC * HBINS * HBINS;   // 98304 == out_size
    zero_kernel<<<(n + 255) / 256, 256, 0, stream>>>(out, n);
    hist_kernel<<<NB * NC * CHUNKS, 256, 0, stream>>>(x, out);
    norm_kernel<<<NB, 256, 0, stream>>>(out);
}
